// ResamplerSD3_16690242912909
// MI455X (gfx1250) — matrix operand order repaired and checked by reading, not compiled, not run on hardware
//
#include <hip/hip_runtime.h>
#include <math.h>

// ---------------------------------------------------------------------------
// Types for CDNA5 WMMA (wave32)
// ---------------------------------------------------------------------------
typedef __bf16 bf16;
typedef __attribute__((ext_vector_type(16))) __bf16       v16bf;
typedef __attribute__((ext_vector_type(8)))  float        v8f;
typedef __attribute__((ext_vector_type(4)))  float        v4f;
typedef __attribute__((ext_vector_type(4)))  unsigned int u32x4;
typedef __attribute__((ext_vector_type(4)))  int          v4i;

// gfx1250 async global->LDS copy path (ASYNCcnt). Builtin confirmed present.
#if defined(__HIP_DEVICE_COMPILE__) && \
    __has_builtin(__builtin_amdgcn_global_load_async_to_lds_b128)
#define USE_ASYNC_COPY 1
#else
#define USE_ASYNC_COPY 0
#endif

__device__ __forceinline__ v8f v8f_zero() {
  v8f z;
#pragma unroll
  for (int i = 0; i < 8; ++i) z[i] = 0.f;
  return z;
}

__device__ __forceinline__ unsigned int pack2(float a, float b) {
  const unsigned int lo = (unsigned int)__builtin_bit_cast(unsigned short, (bf16)a);
  const unsigned int hi = (unsigned int)__builtin_bit_cast(unsigned short, (bf16)b);
  return lo | (hi << 16);
}

// 16B global->LDS copy of 8 bf16 (async: lands in LDS without VGPR roundtrip)
__device__ __forceinline__ void g2l_b128(bf16* dst, const bf16* src) {
#if USE_ASYNC_COPY
  __builtin_amdgcn_global_load_async_to_lds_b128(
      (__attribute__((address_space(1))) v4i*)(src),
      (__attribute__((address_space(3))) v4i*)(dst), 0, 0);
#else
  *(u32x4*)dst = *(const u32x4*)src;
#endif
}

// wait for this wave's outstanding async copies (no-op in sync fallback)
__device__ __forceinline__ void async_join() {
#if USE_ASYNC_COPY
#if __has_builtin(__builtin_amdgcn_s_wait_asynccnt)
  __builtin_amdgcn_s_wait_asynccnt(0);
#else
  asm volatile("s_wait_asynccnt 0x0" ::: "memory");
#endif
#endif
}

// Stage one 16-elem row segment of A into LDS (bf16 raw copy / fp32 convert).
__device__ __forceinline__ void stage_a_row(bf16* dst, const bf16* src) {
  g2l_b128(dst, src);
  g2l_b128(dst + 8, src + 8);
}
__device__ __forceinline__ void stage_a_row(bf16* dst, const float* src) {
  const v4f f0 = *(const v4f*)(src);
  const v4f f1 = *(const v4f*)(src + 4);
  const v4f f2 = *(const v4f*)(src + 8);
  const v4f f3 = *(const v4f*)(src + 12);
  u32x4 q0, q1;
  q0[0] = pack2(f0[0], f0[1]); q0[1] = pack2(f0[2], f0[3]);
  q0[2] = pack2(f1[0], f1[1]); q0[3] = pack2(f1[2], f1[3]);
  q1[0] = pack2(f2[0], f2[1]); q1[1] = pack2(f2[2], f2[3]);
  q1[2] = pack2(f3[0], f3[1]); q1[3] = pack2(f3[2], f3[3]);
  *(u32x4*)dst = q0;
  *(u32x4*)(dst + 8) = q1;
}

// A-fragment (16x32 bf16): lane m = lane&15, k-half = lane>>4.
__device__ __forceinline__ v16bf load_a_frag(const bf16* base, int stride,
                                             int mrow, int koff, int lane) {
  const int m = lane & 15;
  const int kh = (lane >> 4) & 1;
  const bf16* p = base + (mrow + m) * stride + koff + kh * 8;
  union { v16bf v; u32x4 q[2]; } r;
  r.q[0] = *(const u32x4*)(p);
  r.q[1] = *(const u32x4*)(p + 16);
  return r.v;
}

// B-fragment (32x16 bf16) from LDS tile stored TRANSPOSED as Bt[n][k].
__device__ __forceinline__ v16bf load_b_frag(const bf16* baseT, int stride,
                                             int ncol, int koff, int lane) {
  const int n = lane & 15;
  const int kb = (lane >> 4) * 8;
  const bf16* p = baseT + (ncol + n) * stride + koff + kb;
  union { v16bf v; u32x4 q[2]; } r;
  r.q[0] = *(const u32x4*)(p);
  r.q[1] = *(const u32x4*)(p + 16);
  return r.v;
}

__device__ __forceinline__ v8f wmma_bf16(v16bf a, v16bf b, v8f c) {
  return __builtin_amdgcn_wmma_f32_16x16x32_bf16(false, a, false, b,
                                                 (short)0, c, false, false);
}

// ---------------------------------------------------------------------------
// Generic tiled GEMM: C[M,N] = act(A[M,K] @ B[K,N] + bias) + resid
// Block tile 128x128x32, 256 threads (8 waves), wave tile 64x32 (4x2 WMMA).
// Double-buffered LDS: async A copies + packed-B stores for tile k+1 overlap
// the WMMAs on tile k; one s_wait_asynccnt + one barrier per k-step.
// Ragged M handled by clamping the staged row (extra rows never stored).
// ---------------------------------------------------------------------------
template <typename AT, typename OT, bool GELU>
__global__ __launch_bounds__(256) void gemm_wmma(
    const AT* __restrict__ A, int lda,
    const float* __restrict__ B, int ldb,
    const float* __restrict__ bias,
    const float* __restrict__ resid, int ldr,
    OT* __restrict__ C, int ldc,
    int M, int N, int K) {
  constexpr int BM = 128, BN = 128, BK = 32;
  constexpr int SA = 40, SB = 40;  // padded LDS strides (16B-aligned rows)
  __shared__ bf16 As[2][BM * SA];
  __shared__ bf16 Bt[2][BN * SB];

  const int t = threadIdx.x;
  const int lane = t & 31;
  const int w = t >> 5;
  const int wm = w & 1;   // 2 M-halves of 64
  const int wn = w >> 1;  // 4 N-quarters of 32
  const int m0 = blockIdx.y * BM;
  const int n0 = blockIdx.x * BN;

  // A staging address (clamped row): thread covers one 16-elem row segment
  const int arow = t >> 1;
  const int aseg = (t & 1) * 16;
  int agr = m0 + arow;
  if (agr >= M) agr = M - 1;
  const AT* aptr = A + (long)agr * lda + aseg;

  // B staging: 16 strided fp32 loads -> packed bf16 -> two b128 LDS stores
  const int bn = t & 127;
  const int bkh = t >> 7;
  const float* bptr = B + (long)(bkh * 16) * ldb + n0 + bn;

  auto stage = [&](int k0, int buf) {
    stage_a_row(&As[buf][arow * SA + aseg], aptr + k0);
    float f[16];
#pragma unroll
    for (int e = 0; e < 16; ++e) f[e] = bptr[(long)(k0 + e) * ldb];
    u32x4 q0, q1;
    q0[0] = pack2(f[0], f[1]);   q0[1] = pack2(f[2], f[3]);
    q0[2] = pack2(f[4], f[5]);   q0[3] = pack2(f[6], f[7]);
    q1[0] = pack2(f[8], f[9]);   q1[1] = pack2(f[10], f[11]);
    q1[2] = pack2(f[12], f[13]); q1[3] = pack2(f[14], f[15]);
    *(u32x4*)&Bt[buf][bn * SB + bkh * 16] = q0;
    *(u32x4*)&Bt[buf][bn * SB + bkh * 16 + 8] = q1;
  };

  v8f acc[4][2];
#pragma unroll
  for (int i = 0; i < 4; ++i)
#pragma unroll
    for (int j = 0; j < 2; ++j) acc[i][j] = v8f_zero();

  stage(0, 0);
  async_join();
  __syncthreads();

  int cur = 0;
  for (int k0 = 0; k0 < K; k0 += BK) {
    // kick off next tile into the alternate buffer (overlaps with WMMAs)
    if (k0 + BK < K) {
      stage(k0 + BK, cur ^ 1);
      if (k0 + 2 * BK < K)  // gfx1250 global_prefetch_b8 on the tile after
        __builtin_prefetch(bptr + (long)(k0 + 2 * BK) * ldb, 0, 0);
    }
    // compute on current buffer
    const bf16* asb = As[cur];
    const bf16* btb = Bt[cur];
    v16bf bfr[2];
#pragma unroll
    for (int j = 0; j < 2; ++j)
      bfr[j] = load_b_frag(btb, SB, wn * 32 + j * 16, 0, lane);
#pragma unroll
    for (int i = 0; i < 4; ++i) {
      v16bf af = load_a_frag(asb, SA, wm * 64 + i * 16, 0, lane);
#pragma unroll
      for (int j = 0; j < 2; ++j) acc[i][j] = wmma_bf16(af, bfr[j], acc[i][j]);
    }
    async_join();
    __syncthreads();
    cur ^= 1;
  }

  // epilogue: C layout lane n = lane&15, row = r + (lane>>4)*8
#pragma unroll
  for (int i = 0; i < 4; ++i)
#pragma unroll
    for (int j = 0; j < 2; ++j)
#pragma unroll
      for (int r = 0; r < 8; ++r) {
        const int row = m0 + wm * 64 + i * 16 + r + ((lane >> 4) << 3);
        const int col = n0 + wn * 32 + j * 16 + (lane & 15);
        if (row < M) {
          float v = acc[i][j][r];
          if (bias) v += bias[col];
          if (GELU) v = 0.5f * v * (1.f + erff(v * 0.70710678118654752f));
          if (resid) v += resid[(long)row * ldr + col];
          C[(long)row * ldc + col] = (OT)v;
        }
      }
}

// ---------------------------------------------------------------------------
// Flash-style attention, one workgroup per (b, h). NQ=64, DH=64, NK=321.
// ---------------------------------------------------------------------------
__global__ __launch_bounds__(256) void attention_kernel(
    const bf16* __restrict__ q, const bf16* __restrict__ kv,
    bf16* __restrict__ o) {
  constexpr int SQ = 72, SK = 72, SV = 40, SS = 36, SP = 40;
  __shared__ bf16 Qs[64 * SQ];
  __shared__ bf16 Ks[32 * SK];   // K chunk as Bt[kpos][d] (native layout of k)
  __shared__ bf16 Vt[64 * SV];   // V chunk transposed: Vt[d][kpos]
  __shared__ float Sc[64 * SS];  // score chunk fp32
  __shared__ bf16 Pb[64 * SP];   // softmax probs bf16
  __shared__ float rowM[64], rowL[64], rowA[64];

  const int t = threadIdx.x;
  const int lane = t & 31;
  const int w = t >> 5;
  const int mg = w & 3;   // q-row group (16 rows)
  const int dh = w >> 2;  // output col half (32 cols)
  const int b = blockIdx.x >> 4;
  const int h = blockIdx.x & 15;
  const long qbase = (long)(b * 64) * 1024 + h * 64;
  const long kvbase = (long)(b * 321) * 2048 + h * 64;

  // stage Q: raw async b128 copies (scale applied to scores instead)
  {
    const int m = t >> 2;
    const int seg = (t & 3) * 16;
    const bf16* src = q + qbase + (long)m * 1024 + seg;
    stage_a_row(&Qs[m * SQ + seg], src);
  }
  if (t < 64) { rowM[t] = -3.0e38f; rowL[t] = 0.f; }
  v8f oacc[2];
  oacc[0] = v8f_zero();
  oacc[1] = v8f_zero();
  async_join();
  __syncthreads();

  for (int kc = 0; kc < 11; ++kc) {  // 11 chunks of 32 cover 321 (tail masked)
    const int kbase = kc * 32;
    // stage K chunk: one async b128 copy per thread (clamped key row)
    {
      const int kp = t >> 3;
      const int seg = (t & 7) * 8;
      int kpos = kbase + kp;
      if (kpos > 320) kpos = 320;  // tail cols masked to -inf below
      g2l_b128(&Ks[kp * SK + seg], &kv[kvbase + (long)kpos * 2048 + seg]);
    }
    // stage V chunk transposed: vector global load, scalar LDS scatter
    {
      const int kp = t >> 3;
      const int dseg = (t & 7) * 8;
      int kpos = kbase + kp;
      if (kpos > 320) kpos = 320;  // tail weighted by exp(-inf)=0
      union { u32x4 qv; bf16 hx[8]; } uv;
      uv.qv = *(const u32x4*)&kv[kvbase + (long)kpos * 2048 + 1024 + dseg];
#pragma unroll
      for (int e = 0; e < 8; ++e) Vt[(dseg + e) * SV + kp] = uv.hx[e];
    }
    async_join();
    __syncthreads();

    // S = Q @ K^T on waves 0..3 (wave-uniform -> EXEC all ones)
    if (dh == 0) {
      v8f s0 = v8f_zero(), s1 = v8f_zero();
#pragma unroll
      for (int kk = 0; kk < 64; kk += 32) {
        v16bf af = load_a_frag(Qs, SQ, mg * 16, kk, lane);
        v16bf b0 = load_b_frag(Ks, SK, 0, kk, lane);
        v16bf b1 = load_b_frag(Ks, SK, 16, kk, lane);
        s0 = wmma_bf16(af, b0, s0);
        s1 = wmma_bf16(af, b1, s1);
      }
#pragma unroll
      for (int r = 0; r < 8; ++r) {
        const int row = mg * 16 + r + ((lane >> 4) << 3);
        const int c0 = lane & 15;
        float v0 = s0[r] * 0.125f;  // q*scale, k*scale folded: 64^-0.5
        float v1 = s1[r] * 0.125f;
        if (kbase + c0 >= 321) v0 = -1.0e30f;
        if (kbase + 16 + c0 >= 321) v1 = -1.0e30f;
        Sc[row * SS + c0] = v0;
        Sc[row * SS + 16 + c0] = v1;
      }
    }
    __syncthreads();

    // online softmax: threads 0..63 (= waves 0,1) own one row each
    if (t < 64) {
      const float mOld = rowM[t];
      float mNew = mOld;
#pragma unroll
      for (int c = 0; c < 32; ++c) mNew = fmaxf(mNew, Sc[t * SS + c]);
      const float alpha = __expf(mOld - mNew);
      float sum = 0.f;
#pragma unroll
      for (int c = 0; c < 32; ++c) {
        const float p = __expf(Sc[t * SS + c] - mNew);
        sum += p;
        Pb[t * SP + c] = (bf16)p;
      }
      rowM[t] = mNew;
      rowL[t] = rowL[t] * alpha + sum;
      rowA[t] = alpha;
    }
    __syncthreads();

    // rescale running O and accumulate P @ V
#pragma unroll
    for (int j = 0; j < 2; ++j)
#pragma unroll
      for (int r = 0; r < 8; ++r)
        oacc[j][r] *= rowA[mg * 16 + r + ((lane >> 4) << 3)];
    {
      v16bf af = load_a_frag(Pb, SP, mg * 16, 0, lane);
#pragma unroll
      for (int j = 0; j < 2; ++j) {
        v16bf bfb = load_b_frag(Vt, SV, dh * 32 + j * 16, 0, lane);
        oacc[j] = wmma_bf16(af, bfb, oacc[j]);
      }
    }
    __syncthreads();
  }

  // write O (normalized)
#pragma unroll
  for (int j = 0; j < 2; ++j)
#pragma unroll
    for (int r = 0; r < 8; ++r) {
      const int row = mg * 16 + r + ((lane >> 4) << 3);
      const int col = dh * 32 + j * 16 + (lane & 15);
      o[qbase + (long)row * 1024 + col] = (bf16)(oacc[j][r] / rowL[row]);
    }
}

// ---------------------------------------------------------------------------
// LayerNorm over rows of 1024, fp32 in, OT out(s). Output row remapping:
//   mode 0: identity; mode 1: x rows -> kvA (b*321+s); mode 2: lat -> kvA
// ---------------------------------------------------------------------------
__device__ __forceinline__ long ln_orow(int row, int mode) {
  if (mode == 1) { const int b = row / 257, s = row % 257; return (long)b * 321 + s; }
  if (mode == 2) { const int b = row >> 6, s = row & 63; return (long)b * 321 + 257 + s; }
  return row;
}

template <typename OT>
__global__ __launch_bounds__(256) void ln1024_kernel(
    const float* __restrict__ in, const float* __restrict__ g,
    const float* __restrict__ bb, OT* __restrict__ outA, int modeA,
    OT* __restrict__ outB, int modeB) {
  __shared__ float red[256];
  const int row = blockIdx.x;
  const int t = threadIdx.x;
  const float* x = in + (long)row * 1024;
  float v[4];
#pragma unroll
  for (int e = 0; e < 4; ++e) v[e] = x[t + e * 256];
  red[t] = v[0] + v[1] + v[2] + v[3];
  __syncthreads();
  for (int off = 128; off > 0; off >>= 1) {
    if (t < off) red[t] += red[t + off];
    __syncthreads();
  }
  const float mean = red[0] * (1.f / 1024.f);
  __syncthreads();
  float qsum = 0.f;
#pragma unroll
  for (int e = 0; e < 4; ++e) {
    const float d = v[e] - mean;
    qsum += d * d;
  }
  red[t] = qsum;
  __syncthreads();
  for (int off = 128; off > 0; off >>= 1) {
    if (t < off) red[t] += red[t + off];
    __syncthreads();
  }
  const float rs = rsqrtf(red[0] * (1.f / 1024.f) + 1e-5f);
  const long ra = ln_orow(row, modeA) * 1024;
  const long rb = outB ? ln_orow(row, modeB) * 1024 : 0;
#pragma unroll
  for (int e = 0; e < 4; ++e) {
    const int idx = t + e * 256;
    const float y = (v[e] - mean) * rs * g[idx] + bb[idx];
    outA[ra + idx] = (OT)y;
    if (outB) outB[rb + idx] = (OT)y;
  }
}

// broadcast latents (1,64,1024) -> lat (32,64,1024) fp32
__global__ __launch_bounds__(256) void bcast_lat(const float* __restrict__ lin,
                                                 float* __restrict__ lout) {
  const int i = blockIdx.x * 256 + threadIdx.x;
  lout[i] = lin[i & 65535];  // 64*1024 = 65536
}

// ---------------------------------------------------------------------------
// Host-side launch helpers
// ---------------------------------------------------------------------------
template <typename AT, typename OT, bool GELU>
static void launch_gemm(const AT* A, int lda, const float* B, int ldb,
                        const float* bias, const float* resid, int ldr, OT* C,
                        int ldc, int M, int N, int K, hipStream_t stream) {
  dim3 grid(N / 128, (M + 127) / 128);
  gemm_wmma<AT, OT, GELU><<<grid, 256, 0, stream>>>(A, lda, B, ldb, bias, resid,
                                                    ldr, C, ldc, M, N, K);
}

extern "C" void kernel_launch(void* const* d_in, const int* in_sizes, int n_in,
                              void* d_out, int out_size, void* d_ws,
                              size_t ws_size, hipStream_t stream) {
  (void)in_sizes; (void)n_in; (void)out_size; (void)ws_size;
  const float* x          = (const float*)d_in[0];
  const float* latents    = (const float*)d_in[1];
  const float* proj_in_w  = (const float*)d_in[2];
  const float* proj_in_b  = (const float*)d_in[3];
  const float* ln1_g      = (const float*)d_in[4];
  const float* ln1_b      = (const float*)d_in[5];
  const float* ln2_g      = (const float*)d_in[6];
  const float* ln2_b      = (const float*)d_in[7];
  const float* Wq         = (const float*)d_in[8];
  const float* Wkv        = (const float*)d_in[9];
  const float* Wo         = (const float*)d_in[10];
  const float* ff_ln_g    = (const float*)d_in[11];
  const float* ff_ln_b    = (const float*)d_in[12];
  const float* ff_w1      = (const float*)d_in[13];
  const float* ff_w2      = (const float*)d_in[14];
  const float* proj_out_w = (const float*)d_in[15];
  const float* proj_out_b = (const float*)d_in[16];
  const float* norm_out_g = (const float*)d_in[17];
  const float* norm_out_b = (const float*)d_in[18];
  float* out = (float*)d_out;
  char* ws = (char*)d_ws;

  const int MX = 32 * 257;   // 8224 rows of x
  const int ML = 32 * 64;    // 2048 latent rows
  const int MKV = 32 * 321;  // 10272 concatenated rows

  size_t off = 0;
  auto take = [&](size_t bytes) {
    size_t r = off;
    off += (bytes + 255) & ~(size_t)255;
    return r;
  };
  float* xf   = (float*)(ws + take((size_t)MX * 1024 * 4));   // proj_in out
  float* latf = (float*)(ws + take((size_t)ML * 1024 * 4));   // running lat
  bf16*  latn = (bf16*)(ws + take((size_t)ML * 1024 * 2));    // LN(lat)
  bf16*  kvA  = (bf16*)(ws + take((size_t)MKV * 1024 * 2));   // [xn; latn]
  bf16*  qb   = (bf16*)(ws + take((size_t)ML * 1024 * 2));    // Q
  bf16*  kvb  = (bf16*)(ws + take((size_t)MKV * 2048 * 2));   // K|V
  bf16*  ob   = (bf16*)(ws + take((size_t)ML * 1024 * 2));    // attn out
  bf16*  hn   = (bf16*)(ws + take((size_t)ML * 1024 * 2));    // FF LN
  bf16*  h1   = (bf16*)(ws + take((size_t)ML * 4096 * 2));    // gelu(h@W1)
  float* tmp  = (float*)(ws + take((size_t)ML * 1024 * 4));   // proj_out

  // lat = broadcast(latents)
  bcast_lat<<<(ML * 1024) / 256, 256, 0, stream>>>(latents, latf);
  // xf = x @ proj_in_w + b
  launch_gemm<float, float, false>(x, 768, proj_in_w, 1024, proj_in_b, nullptr,
                                   0, xf, 1024, MX, 1024, 768, stream);

  for (int i = 0; i < 8; ++i) {
    // xn -> kvA[:, 0:257], (latn, kvA[:, 257:321]) = LN(lat)
    ln1024_kernel<bf16><<<MX, 256, 0, stream>>>(
        xf, ln1_g + i * 1024, ln1_b + i * 1024, kvA, 1, (bf16*)nullptr, 0);
    ln1024_kernel<bf16><<<ML, 256, 0, stream>>>(
        latf, ln2_g + i * 1024, ln2_b + i * 1024, latn, 0, kvA, 2);
    // Q = latn @ Wq[i]
    launch_gemm<bf16, bf16, false>(latn, 1024, Wq + (size_t)i * 1024 * 1024,
                                   1024, nullptr, nullptr, 0, qb, 1024, ML,
                                   1024, 1024, stream);
    // KV = [xn; latn] @ Wkv[i]
    launch_gemm<bf16, bf16, false>(kvA, 1024, Wkv + (size_t)i * 1024 * 2048,
                                   2048, nullptr, nullptr, 0, kvb, 2048, MKV,
                                   2048, 1024, stream);
    // attention per (b, h)
    attention_kernel<<<512, 256, 0, stream>>>(qb, kvb, ob);
    // lat = o @ Wo[i] + lat
    launch_gemm<bf16, float, false>(ob, 1024, Wo + (size_t)i * 1024 * 1024,
                                    1024, nullptr, latf, 1024, latf, 1024, ML,
                                    1024, 1024, stream);
    // FF: hn = LN(lat); h1 = gelu(hn @ W1); lat = h1 @ W2 + lat
    ln1024_kernel<bf16><<<ML, 256, 0, stream>>>(
        latf, ff_ln_g + i * 1024, ff_ln_b + i * 1024, hn, 0, (bf16*)nullptr, 0);
    launch_gemm<bf16, bf16, true>(hn, 1024, ff_w1 + (size_t)i * 1024 * 4096,
                                  4096, nullptr, nullptr, 0, h1, 4096, ML, 4096,
                                  1024, stream);
    launch_gemm<bf16, float, false>(h1, 4096, ff_w2 + (size_t)i * 4096 * 1024,
                                    1024, nullptr, latf, 1024, latf, 1024, ML,
                                    1024, 4096, stream);
  }

  // out = LN(lat @ proj_out_w + b)
  launch_gemm<float, float, false>(latf, 1024, proj_out_w, 1024, proj_out_b,
                                   nullptr, 0, tmp, 1024, ML, 1024, 1024,
                                   stream);
  ln1024_kernel<float><<<ML, 256, 0, stream>>>(tmp, norm_out_g, norm_out_b, out,
                                               0, (float*)nullptr, 0);
}
